// SentenceVAE_47991964566213
// MI455X (gfx1250) — matrix operand order repaired and checked by reading, not compiled, not run on hardware
//
#include <hip/hip_runtime.h>
#include <hip/hip_bf16.h>
#include <math.h>

typedef __bf16 bf16;
typedef __attribute__((ext_vector_type(16))) __bf16 v16bf;
typedef __attribute__((ext_vector_type(8)))  __bf16 v8bf;
typedef __attribute__((ext_vector_type(8)))  float  v8f;
typedef __attribute__((ext_vector_type(4)))  unsigned int u32x4;
typedef __attribute__((ext_vector_type(8)))  int i32x8;
typedef __attribute__((ext_vector_type(4)))  int i32x4;

#define VV 32000
#define EE 256
#define HH 512
#define ZZ 64
#define BB 32
#define TT 128
#define H4 (4*HH)
#define BT (BB*TT)

#if defined(__gfx1250__) && __has_builtin(__builtin_amdgcn_tensor_load_to_lds) && \
    __has_builtin(__builtin_amdgcn_s_wait_tensorcnt)
#define USE_TDM 1
#else
#define USE_TDM 0
#endif

// ---------------------------------------------------------------------------
// WMMA fragment helpers (gfx1250 wave32; A/B use contiguous k = 16*(lane/16) + s, C/D per CDNA5 ISA §7.12.2)
// ---------------------------------------------------------------------------

// A matrix 16x32 bf16, row-major source with leading dim lda.
// lane<16 : row M=lane,    K = k0+0..15
// lane>=16: row M=lane-16, K = k0+16..31
__device__ __forceinline__ v16bf load_frag_a(const bf16* __restrict__ A, int lda,
                                             int row0, int k0) {
    int lane = threadIdx.x & 31;
    int m  = row0 + (lane & 15);
    int kk = k0 + ((lane >> 4) << 4);
    const bf16* p = A + (size_t)m * lda + kk;
    union { v16bf v; v8bf h[2]; } u;
    u.h[0] = *(const v8bf*)(p);
    u.h[1] = *(const v8bf*)(p + 8);
    return u.v;
}

// B matrix 32x16 (KxN) taken from row-major weight W[N][K] (so C = A @ W^T).
// lane<16 : col N=lane,    K = k0+0..15
// lane>=16: col N=lane-16, K = k0+16..31
__device__ __forceinline__ v16bf load_frag_b(const bf16* __restrict__ W, int ldw,
                                             int col0, int k0) {
    int lane = threadIdx.x & 31;
    int n  = col0 + (lane & 15);
    int kk = k0 + ((lane >> 4) << 4);
    const bf16* p = W + (size_t)n * ldw + kk;
    union { v16bf v; v8bf h[2]; } u;
    u.h[0] = *(const v8bf*)(p);
    u.h[1] = *(const v8bf*)(p + 8);
    return u.v;
}

// Same fragment gather but from an LDS-resident 16xK tile (rows relative to tile).
__device__ __forceinline__ v16bf load_frag_b_lds(const bf16* tile, int ldw, int k0) {
    int lane = threadIdx.x & 31;
    int n  = lane & 15;
    int kk = k0 + ((lane >> 4) << 4);
    const bf16* p = tile + (size_t)n * ldw + kk;
    union { v16bf v; v8bf h[2]; } u;
    u.h[0] = *(const v8bf*)(p);
    u.h[1] = *(const v8bf*)(p + 8);
    return u.v;
}

// C/D 16x16 f32: lane = N + (M>=8 ? 16 : 0), VGPR r = M%8.
__device__ __forceinline__ void store_tile(float* __restrict__ C, int ldc,
                                           int row0, int col0, v8f acc,
                                           const float* __restrict__ bias) {
    int lane  = threadIdx.x & 31;
    int n     = col0 + (lane & 15);
    int mbase = row0 + ((lane >> 4) << 3);
    float bv = bias ? bias[n] : 0.0f;
#pragma unroll
    for (int r = 0; r < 8; ++r)
        C[(size_t)(mbase + r) * ldc + n] = acc[r] + bv;
}

__device__ __forceinline__ float sigm(float x) { return 1.0f / (1.0f + __expf(-x)); }

#if USE_TDM
// Issue a TDM load of a 16-row x 512-col bf16 tile (row-major, stride 512)
// from global memory into LDS. D# per CDNA5 ISA.
__device__ __forceinline__ void tdm_load_tile(const bf16* gptr, unsigned lds_off) {
    unsigned long long ga = (unsigned long long)(uintptr_t)gptr;
    u32x4 g0;
    g0[0] = 1u;                                           // count=1, user mode
    g0[1] = lds_off;                                      // lds_addr (bytes)
    g0[2] = (unsigned)(ga & 0xffffffffu);                 // global_addr[31:0]
    g0[3] = (unsigned)((ga >> 32) & 0x01ffffffu)          // global_addr[56:32]
          | (2u << 30);                                   // type=2 (image)
    i32x8 g1;
    g1[0] = (int)(1u << 16);                              // data_size=1 -> 2 bytes
    g1[1] = (int)(((unsigned)HH & 0xffffu) << 16);        // tensor_dim0[15:0]=512
    g1[2] = (int)(16u << 16);                             // tensor_dim1[15:0]=16
    g1[3] = (int)(((unsigned)HH) << 16);                  // tile_dim0=512
    g1[4] = 16;                                           // tile_dim1=16
    g1[5] = HH;                                           // tensor_dim0_stride=512
    g1[6] = 0;
    g1[7] = 0;
    i32x4 z4 = {0, 0, 0, 0};
#if __clang_major__ >= 23
    i32x8 z8 = {0, 0, 0, 0, 0, 0, 0, 0};
    __builtin_amdgcn_tensor_load_to_lds(g0, g1, z4, z4, z8, 0);
#else
    __builtin_amdgcn_tensor_load_to_lds(g0, g1, z4, z4, 0);
#endif
}
#endif

// ---------------------------------------------------------------------------
// Generic bf16 WMMA GEMM: C[M,N] = A[M,K] @ W[N,K]^T (+bias), f32 accumulate.
// One wave per block; each wave owns a 16x64 strip of C (4 accumulators).
// grid = (N/64, M/16), block = 32.
// ---------------------------------------------------------------------------
__global__ __launch_bounds__(32)
void wmma_gemm_bf16(const bf16* __restrict__ A, const bf16* __restrict__ W,
                    float* __restrict__ C, const float* __restrict__ bias,
                    int N, int K) {
    int row0 = blockIdx.y * 16;
    int col0 = blockIdx.x * 64;
    v8f acc[4] = {v8f{}, v8f{}, v8f{}, v8f{}};
    for (int k = 0; k < K; k += 32) {
        v16bf a = load_frag_a(A, K, row0, k);
#pragma unroll
        for (int j = 0; j < 4; ++j) {
            v16bf b = load_frag_b(W, K, col0 + 16 * j, k);
            acc[j] = __builtin_amdgcn_wmma_f32_16x16x32_bf16(
                false, a, false, b, (short)0, acc[j], false, false);
        }
    }
#pragma unroll
    for (int j = 0; j < 4; ++j)
        store_tile(C, N, row0, col0 + 16 * j, acc[j], bias);
}

// ---------------------------------------------------------------------------
// Elementwise / gather kernels
// ---------------------------------------------------------------------------
__global__ void gather_emb_kernel(const float* __restrict__ emb, const int* __restrict__ x,
                                  bf16* __restrict__ out, int reverse) {
    int idx = blockIdx.x * blockDim.x + threadIdx.x;
    if (idx >= BT * EE) return;
    int tok = idx / EE, e = idx % EE;
    int b = tok / TT, t = tok % TT;
    int src = reverse ? x[b * TT + (TT - 1 - t)] : x[tok];
    out[idx] = (bf16)emb[(size_t)src * EE + e];
}

__global__ void cvt_bf16_kernel(const float* __restrict__ in, bf16* __restrict__ out, int n) {
    int idx = blockIdx.x * blockDim.x + threadIdx.x;
    if (idx < n) out[idx] = (bf16)in[idx];
}

__global__ void zero_f32_kernel(float* __restrict__ p, int n) {
    int idx = blockIdx.x * blockDim.x + threadIdx.x;
    if (idx < n) p[idx] = 0.0f;
}

// LSTM gate fusion: g = gih[b*T+t] + gtmp[b] + bias; update c,h; emit bf16 h.
__global__ void lstm_gate_kernel(const float* __restrict__ gih, int t,
                                 const float* __restrict__ gtmp,
                                 const float* __restrict__ bias,
                                 float* __restrict__ h, float* __restrict__ c,
                                 bf16* __restrict__ h_bf, bf16* __restrict__ hs_bf) {
    int idx = blockIdx.x * blockDim.x + threadIdx.x;
    if (idx >= BB * HH) return;
    int b = idx / HH, j = idx % HH;
    const float* grow = gih + (size_t)(b * TT + t) * H4;
    const float* trow = gtmp + (size_t)b * H4;
    float gi = grow[j]          + trow[j]          + bias[j];
    float gf = grow[HH + j]     + trow[HH + j]     + bias[HH + j];
    float gg = grow[2 * HH + j] + trow[2 * HH + j] + bias[2 * HH + j];
    float go = grow[3 * HH + j] + trow[3 * HH + j] + bias[3 * HH + j];
    float cn = sigm(gf) * c[idx] + sigm(gi) * tanhf(gg);
    float hn = sigm(go) * tanhf(cn);
    c[idx] = cn;
    h[idx] = hn;
    h_bf[idx] = (bf16)hn;
    if (hs_bf) hs_bf[(size_t)(b * TT + t) * HH + j] = (bf16)hn;
}

__global__ void concat_bf16_kernel(const float* __restrict__ hf, const float* __restrict__ hb,
                                   bf16* __restrict__ out) {
    int idx = blockIdx.x * blockDim.x + threadIdx.x;
    if (idx >= BB * 2 * HH) return;
    int b = idx / (2 * HH), j = idx % (2 * HH);
    float v = (j < HH) ? hf[b * HH + j] : hb[b * HH + (j - HH)];
    out[idx] = (bf16)v;
}

// sigma = softplus(sraw); z = mu + sigma*eps  (eps broadcast over batch)
__global__ void vae_reparam_kernel(const float* __restrict__ mu, float* __restrict__ sig,
                                   const float* __restrict__ eps, bf16* __restrict__ z_bf) {
    int idx = blockIdx.x * blockDim.x + threadIdx.x;
    if (idx >= BB * ZZ) return;
    float x = sig[idx];
    float sp = (x > 20.0f) ? x : log1pf(__expf(x));
    sig[idx] = sp;
    float z = mu[idx] + sp * eps[idx % ZZ];
    z_bf[idx] = (bf16)z;
}

__global__ void tanh_h0_kernel(float* __restrict__ h, bf16* __restrict__ h_bf) {
    int idx = blockIdx.x * blockDim.x + threadIdx.x;
    if (idx >= BB * HH) return;
    float v = tanhf(h[idx]);
    h[idx] = v;
    h_bf[idx] = (bf16)v;
}

__global__ void kl_kernel(const float* __restrict__ mu, const float* __restrict__ sig,
                          float* __restrict__ out) {
    int idx = blockIdx.x * blockDim.x + threadIdx.x;
    if (idx >= BB * ZZ) return;
    float m = mu[idx], s = sig[idx];
    atomicAdd(out, -0.5f * (1.0f + __logf(s) - m * m - s));
}

__global__ void scale_kernel(float* __restrict__ out) {
    if (threadIdx.x == 0 && blockIdx.x == 0) out[0] *= (1.0f / (float)BB);
}

// ---------------------------------------------------------------------------
// Fused output projection + online log-softmax + cross-entropy.
// One wave per 16 rows of hs; loops over V=32000 in 16-col WMMA tiles keeping
// a flash-style running (max, sumexp) per row, never materializing logits.
// out_W tiles are staged through LDS by the Tensor Data Mover with double
// buffering (TENSORcnt-synchronized); falls back to direct global loads if the
// TDM builtin is unavailable. grid = BT/16 = 256, block = 32 (one wave).
// ---------------------------------------------------------------------------
__global__ __launch_bounds__(32)
void ce_proj_kernel(const bf16* __restrict__ hs, const bf16* __restrict__ Wout,
                    const float* __restrict__ bout, const int* __restrict__ y,
                    float* __restrict__ out) {
    int lane = threadIdx.x & 31;
    int row0 = blockIdx.x * 16;
    int nl = lane & 15;
    int mbase = (lane >> 4) << 3;

    // Keep all 16 A fragments (16 rows x K=512) resident in VGPRs.
    v16bf afr[16];
#pragma unroll
    for (int kk = 0; kk < 16; ++kk) afr[kk] = load_frag_a(hs, HH, row0, kk * 32);

    float rmax[8], rsum[8], tgt[8];
    int yr[8];
#pragma unroll
    for (int r = 0; r < 8; ++r) {
        rmax[r] = -1e30f; rsum[r] = 0.0f; tgt[r] = 0.0f;
        yr[r] = y[row0 + mbase + r];
    }

    const int NTILE = VV / 16;   // 2000

#if USE_TDM
    __shared__ __align__(16) bf16 smem[2][16 * HH];      // 2 x 16KB double buffer
    unsigned lds0 = (unsigned)(uintptr_t)&smem[0][0];
    unsigned lds1 = (unsigned)(uintptr_t)&smem[1][0];
    // prime the pipeline with tile 0
    tdm_load_tile(Wout, lds0);
#endif

    for (int tile = 0; tile < NTILE; ++tile) {
        int col0 = tile * 16;
#if USE_TDM
        // issue next tile's DMA into the other buffer, then wait for current
        if (tile + 1 < NTILE) {
            tdm_load_tile(Wout + (size_t)(col0 + 16) * HH,
                          ((tile + 1) & 1) ? lds1 : lds0);
            __builtin_amdgcn_s_wait_tensorcnt(1);
        } else {
            __builtin_amdgcn_s_wait_tensorcnt(0);
        }
        const bf16* wt = &smem[tile & 1][0];
#endif
        v8f acc = {};
#pragma unroll
        for (int kk = 0; kk < 16; ++kk) {
#if USE_TDM
            v16bf b = load_frag_b_lds(wt, HH, kk * 32);
#else
            v16bf b = load_frag_b(Wout, HH, col0, kk * 32);
#endif
            acc = __builtin_amdgcn_wmma_f32_16x16x32_bf16(
                false, afr[kk], false, b, (short)0, acc, false, false);
        }
        float bv = bout[col0 + nl];
        int cn = col0 + nl;
#pragma unroll
        for (int r = 0; r < 8; ++r) {
            float v = acc[r] + bv;
            if (cn == yr[r]) tgt[r] = v;
            float m = rmax[r];
            if (v > m) { rsum[r] = rsum[r] * __expf(m - v) + 1.0f; rmax[r] = v; }
            else       { rsum[r] += __expf(v - m); }
        }
    }

    // Merge the 16 lanes of each half-wave (each lane saw a disjoint column set).
#pragma unroll
    for (int r = 0; r < 8; ++r) {
#pragma unroll
        for (int s = 1; s < 16; s <<= 1) {
            float om = __shfl_xor(rmax[r], s, 32);
            float os = __shfl_xor(rsum[r], s, 32);
            float ot = __shfl_xor(tgt[r],  s, 32);
            float m = fmaxf(rmax[r], om);
            rsum[r] = rsum[r] * __expf(rmax[r] - m) + os * __expf(om - m);
            rmax[r] = m;
            tgt[r] += ot;
        }
    }
    if (nl == 0) {
        float local = 0.0f;
#pragma unroll
        for (int r = 0; r < 8; ++r)
            local += (rmax[r] + __logf(rsum[r])) - tgt[r];
        atomicAdd(out, local);
    }
}

// ---------------------------------------------------------------------------
// Host orchestration
// ---------------------------------------------------------------------------
extern "C" void kernel_launch(void* const* d_in, const int* in_sizes, int n_in,
                              void* d_out, int out_size, void* d_ws, size_t ws_size,
                              hipStream_t stream) {
    const int*   x       = (const int*)d_in[0];
    const int*   y       = (const int*)d_in[1];
    const float* eps     = (const float*)d_in[2];
    const float* enc_emb = (const float*)d_in[3];
    const float* dec_emb = (const float*)d_in[4];
    const float* fwd_Wih = (const float*)d_in[5];
    const float* fwd_Whh = (const float*)d_in[6];
    const float* fwd_b   = (const float*)d_in[7];
    const float* bwd_Wih = (const float*)d_in[8];
    const float* bwd_Whh = (const float*)d_in[9];
    const float* bwd_b   = (const float*)d_in[10];
    const float* dec_Wih = (const float*)d_in[11];
    const float* dec_Whh = (const float*)d_in[12];
    const float* dec_b   = (const float*)d_in[13];
    const float* enc_hb  = (const float*)d_in[15];
    const float* mu_b    = (const float*)d_in[17];
    const float* std_b   = (const float*)d_in[19];
    const float* dec_hb  = (const float*)d_in[21];
    const float* out_b   = (const float*)d_in[23];

    char* ws = (char*)d_ws;
    size_t off = 0;
    auto alloc = [&](size_t bytes) -> void* {
        off = (off + 255) & ~(size_t)255;
        void* p = ws + off;
        off += bytes;
        return p;
    };

    // bf16 activations / weights
    bf16* ef       = (bf16*)alloc((size_t)BT * EE * 2);
    bf16* eb       = (bf16*)alloc((size_t)BT * EE * 2);
    bf16* ed       = (bf16*)alloc((size_t)BT * EE * 2);
    bf16* w_fwd_ih = (bf16*)alloc((size_t)H4 * EE * 2);
    bf16* w_fwd_hh = (bf16*)alloc((size_t)H4 * HH * 2);
    bf16* w_bwd_ih = (bf16*)alloc((size_t)H4 * EE * 2);
    bf16* w_bwd_hh = (bf16*)alloc((size_t)H4 * HH * 2);
    bf16* w_dec_ih = (bf16*)alloc((size_t)H4 * EE * 2);
    bf16* w_dec_hh = (bf16*)alloc((size_t)H4 * HH * 2);
    bf16* w_enc_h  = (bf16*)alloc((size_t)HH * 2 * HH * 2);
    bf16* w_mu     = (bf16*)alloc((size_t)ZZ * HH * 2);
    bf16* w_std    = (bf16*)alloc((size_t)ZZ * HH * 2);
    bf16* w_dec_h  = (bf16*)alloc((size_t)HH * ZZ * 2);
    bf16* w_out    = (bf16*)alloc((size_t)VV * HH * 2);
    bf16* hs_bf    = (bf16*)alloc((size_t)BT * HH * 2);
    bf16* hcat_bf  = (bf16*)alloc((size_t)BB * 2 * HH * 2);
    bf16* henc_bf  = (bf16*)alloc((size_t)BB * HH * 2);
    bf16* z_bf     = (bf16*)alloc((size_t)BB * ZZ * 2);
    bf16* h_fwd_bf = (bf16*)alloc((size_t)BB * HH * 2);
    bf16* h_bwd_bf = (bf16*)alloc((size_t)BB * HH * 2);
    bf16* h_dec_bf = (bf16*)alloc((size_t)BB * HH * 2);

    // f32 buffers
    float* gih_fwd  = (float*)alloc((size_t)BT * H4 * 4);   // reused for decoder
    float* gih_bwd  = (float*)alloc((size_t)BT * H4 * 4);
    float* gtmp_f   = (float*)alloc((size_t)BB * H4 * 4);
    float* gtmp_b   = (float*)alloc((size_t)BB * H4 * 4);
    float* h_fwd    = (float*)alloc((size_t)BB * HH * 4);
    float* c_fwd    = (float*)alloc((size_t)BB * HH * 4);
    float* h_bwd    = (float*)alloc((size_t)BB * HH * 4);
    float* c_bwd    = (float*)alloc((size_t)BB * HH * 4);
    float* h_dec    = (float*)alloc((size_t)BB * HH * 4);
    float* c_dec    = (float*)alloc((size_t)BB * HH * 4);
    float* henc     = (float*)alloc((size_t)BB * HH * 4);
    float* mu       = (float*)alloc((size_t)BB * ZZ * 4);
    float* sig      = (float*)alloc((size_t)BB * ZZ * 4);
    float* gih_dec  = gih_fwd;   // alias: encoder ih-gates dead by decoder time

    float* out = (float*)d_out;

    auto grid1 = [](int n) { return dim3((n + 255) / 256); };
    auto cvt = [&](const float* src, bf16* dst, int n) {
        cvt_bf16_kernel<<<grid1(n), 256, 0, stream>>>(src, dst, n);
    };
    auto gemm = [&](const bf16* A, const bf16* W, float* C, const float* bias,
                    int M, int N, int K) {
        dim3 g(N / 64, M / 16);
        wmma_gemm_bf16<<<g, 32, 0, stream>>>(A, W, C, bias, N, K);
    };
    auto zero = [&](float* p, int n) {
        zero_f32_kernel<<<grid1(n), 256, 0, stream>>>(p, n);
    };

    // 0) init output + LSTM state
    zero(out, out_size);
    zero(h_fwd, BB * HH); zero(c_fwd, BB * HH);
    zero(h_bwd, BB * HH); zero(c_bwd, BB * HH);
    zero(c_dec, BB * HH);
    zero((float*)h_fwd_bf, BB * HH / 2);
    zero((float*)h_bwd_bf, BB * HH / 2);

    // 1) weight conversion f32 -> bf16
    cvt(fwd_Wih, w_fwd_ih, H4 * EE);
    cvt(fwd_Whh, w_fwd_hh, H4 * HH);
    cvt(bwd_Wih, w_bwd_ih, H4 * EE);
    cvt(bwd_Whh, w_bwd_hh, H4 * HH);
    cvt(dec_Wih, w_dec_ih, H4 * EE);
    cvt(dec_Whh, w_dec_hh, H4 * HH);
    cvt((const float*)d_in[14], w_enc_h, HH * 2 * HH);
    cvt((const float*)d_in[16], w_mu,  ZZ * HH);
    cvt((const float*)d_in[18], w_std, ZZ * HH);
    cvt((const float*)d_in[20], w_dec_h, HH * ZZ);
    cvt((const float*)d_in[22], w_out, VV * HH);

    // 2) embedding gathers (bf16)
    gather_emb_kernel<<<grid1(BT * EE), 256, 0, stream>>>(enc_emb, x, ef, 0);
    gather_emb_kernel<<<grid1(BT * EE), 256, 0, stream>>>(enc_emb, x, eb, 1);
    gather_emb_kernel<<<grid1(BT * EE), 256, 0, stream>>>(dec_emb, x, ed, 0);

    // 3) hoisted input-gate GEMMs for the two encoder LSTMs
    gemm(ef, w_fwd_ih, gih_fwd, nullptr, BT, H4, EE);
    gemm(eb, w_bwd_ih, gih_bwd, nullptr, BT, H4, EE);

    // 4) encoder recurrence (fwd + bwd interleaved)
    for (int t = 0; t < TT; ++t) {
        gemm(h_fwd_bf, w_fwd_hh, gtmp_f, nullptr, BB, H4, HH);
        gemm(h_bwd_bf, w_bwd_hh, gtmp_b, nullptr, BB, H4, HH);
        lstm_gate_kernel<<<grid1(BB * HH), 256, 0, stream>>>(
            gih_fwd, t, gtmp_f, fwd_b, h_fwd, c_fwd, h_fwd_bf, nullptr);
        lstm_gate_kernel<<<grid1(BB * HH), 256, 0, stream>>>(
            gih_bwd, t, gtmp_b, bwd_b, h_bwd, c_bwd, h_bwd_bf, nullptr);
    }

    // 5) VAE head
    concat_bf16_kernel<<<grid1(BB * 2 * HH), 256, 0, stream>>>(h_fwd, h_bwd, hcat_bf);
    gemm(hcat_bf, w_enc_h, henc, enc_hb, BB, HH, 2 * HH);
    cvt(henc, henc_bf, BB * HH);
    gemm(henc_bf, w_mu,  mu,  mu_b,  BB, ZZ, HH);
    gemm(henc_bf, w_std, sig, std_b, BB, ZZ, HH);
    vae_reparam_kernel<<<grid1(BB * ZZ), 256, 0, stream>>>(mu, sig, eps, z_bf);
    gemm(z_bf, w_dec_h, h_dec, dec_hb, BB, HH, ZZ);
    tanh_h0_kernel<<<grid1(BB * HH), 256, 0, stream>>>(h_dec, h_dec_bf);

    // 6) decoder: hoisted input gates (reusing gih_fwd memory), then recurrence
    gemm(ed, w_dec_ih, gih_dec, nullptr, BT, H4, EE);
    for (int t = 0; t < TT; ++t) {
        gemm(h_dec_bf, w_dec_hh, gtmp_f, nullptr, BB, H4, HH);
        lstm_gate_kernel<<<grid1(BB * HH), 256, 0, stream>>>(
            gih_dec, t, gtmp_f, dec_b, h_dec, c_dec, h_dec_bf, hs_bf);
    }

    // 7) fused output projection + online log-softmax + CE (TDM double-buffered)
    ce_proj_kernel<<<dim3(BT / 16), 32, 0, stream>>>(hs_bf, w_out, out_b, y, out);

    // 8) KL + final (kl + ce) / B
    kl_kernel<<<grid1(BB * ZZ), 256, 0, stream>>>(mu, sig, out);
    scale_kernel<<<1, 32, 0, stream>>>(out);
}
